// DVAEncoder_15848429322572
// MI455X (gfx1250) — matrix operand order repaired and checked by reading, not compiled, not run on hardware
//
#include <hip/hip_runtime.h>
#include <hip/hip_bf16.h>

typedef __attribute__((ext_vector_type(16))) _Float16 v16h;
typedef __attribute__((ext_vector_type(8)))  _Float16 v8h;
typedef __attribute__((ext_vector_type(8)))  float    v8f;

#define HS   501
#define NVT  3
#define NET  2
#define VS   503
#define BB   32
#define MAXN 48
#define KNB  4
#define NROUNDS 26
#define HSP  512          // padded HS / VS
#define N3P  1536         // padded 3*HS
#define NT_SQ 32          // N-tiles in Wg/Wm  (512/16)
#define NT_WH 96          // N-tiles in whh.T  (1536/16)
#define KSTEPS 16         // K-steps (512/32)

struct P {
    // inputs
    const int *v_type, *pred_idx, *pred_et, *succ_idx, *succ_et;
    const unsigned char *pred_mask, *succ_mask;
    const float *wih[2], *whh[2], *bih[2], *bhh[2];
    const float *Wg[2], *bgv[2], *Wm[2];
    const float *Wu, *bu, *gu, *beu, *Wc1, *bc1, *gc, *bec, *Wc2, *bc2v;
    // workspace
    float     *H32[2];     // [B, MAXN, HSP] f32 master state
    _Float16  *H16[2];     // f16 mirror (tail cols 501..511 stay zero)
    _Float16  *Wg16[2];    // swizzled fragment-major [NT_SQ][KSTEPS][32][16]
    _Float16  *Wm16[2];    // swizzled fragment-major
    _Float16  *Whh16[2];   // swizzled fragment-major [NT_WH][KSTEPS][32][16]
    float     *Hin32[2];   // [B, HSP]
    _Float16  *Hin16[2];   // [B, HSP]
    float     *GH[2];      // [B, N3P]
    float     *Hg;         // [B, HS]
    float     *h1;         // [B, 2*HS]
};

// ---------------- WMMA helpers (CDNA5 wave32 layouts) ----------------

__device__ __forceinline__ v8f wmma_f16(v16h a, v16h b, v8f c) {
    return __builtin_amdgcn_wmma_f32_16x16x32_f16(
        false, a, false, b, (short)0, c, false, false);
}

// Combine two aligned 8-half chunks into a v16h (forces 2x global_load_b128).
__device__ __forceinline__ v16h ld2x8(const _Float16* lo, const _Float16* hi) {
    v8h a = *(const v8h*)lo;
    v8h b = *(const v8h*)hi;
    return __builtin_shufflevector(a, b, 0, 1, 2, 3, 4, 5, 6, 7,
                                         8, 9, 10, 11, 12, 13, 14, 15);
}

// A-matrix 16x32 f16 fragment from a padded row pointer (row of this lane's M).
// a[0..7]  = row[kbase + g*8 .. +7],  a[8..15] = row[kbase + 16 + g*8 .. +7]
__device__ __forceinline__ v16h load_A_row(const _Float16* row, int kbase, int g) {
    return ld2x8(row + kbase + g * 8, row + kbase + 16 + g * 8);
}

// Swizzled B fragment: 16 contiguous halves per lane, coalesced across wave.
__device__ __forceinline__ v16h load_B_sw(const _Float16* Bs, int nt, int ks, int lane) {
    const _Float16* base = Bs + (((size_t)nt * KSTEPS + ks) * 32 + lane) * 16;
    return ld2x8(base, base + 8);
}

__device__ __forceinline__ float sigm(float x) { return 1.0f / (1.0f + expf(-x)); }

// ---------------- setup kernels ----------------

__global__ void k_init(P p) {
    const size_t per = (size_t)BB * MAXN * HSP;
    for (size_t i = (size_t)blockIdx.x * blockDim.x + threadIdx.x;
         i < 2 * per; i += (size_t)gridDim.x * blockDim.x) {
        const int d = (int)(i / per);
        const size_t r = i % per;
        p.H32[d][r] = 0.0f;
        p.H16[d][r] = (_Float16)0.0f;
    }
}

// Swizzle weights into fragment-major f16 layout.
// dest index r: e = r&15, lane = (r>>4)&31, ks = (r>>9)&15, nt = r>>13
// source k = ks*32 + (e<8 ? (lane>>4)*8 + e : 16 + (lane>>4)*8 + e-8) ; n = nt*16 + (lane&15)
__global__ void k_prep(P p) {
    const size_t SQ  = (size_t)HSP * HSP;   // Wg / Wm swizzled element count
    const size_t WH  = (size_t)HSP * N3P;   // whh.T swizzled element count
    const size_t per = 2 * SQ + WH;
    for (size_t i = (size_t)blockIdx.x * blockDim.x + threadIdx.x;
         i < 2 * per; i += (size_t)gridDim.x * blockDim.x) {
        const int d = (int)(i / per);
        size_t r = i % per;
        const float* src;
        _Float16* dst;
        bool isWhh = false;
        if (r < SQ)            { dst = p.Wg16[d]; src = p.Wg[d]; }
        else if (r < 2 * SQ)   { r -= SQ; dst = p.Wm16[d]; src = p.Wm[d]; }
        else                   { r -= 2 * SQ; dst = p.Whh16[d]; src = p.whh[d]; isWhh = true; }
        const int e    = (int)(r & 15);
        const int lane = (int)((r >> 4) & 31);
        const int ks   = (int)((r >> 9) & 15);
        const int nt   = (int)(r >> 13);
        const int k = ks * 32 + (e < 8 ? ((lane >> 4) & 1) * 8 + e : 16 + ((lane >> 4) & 1) * 8 + (e - 8));
        const int n = nt * 16 + (lane & 15);
        float v = 0.0f;
        if (!isWhh) { if (k < VS && n < HS)     v = src[k * HS + n];           }
        else        { if (k < HS && n < 3 * HS) v = src[(size_t)n * HS + k];   }  // whh.T
        dst[r] = (_Float16)v;
    }
}

// ---------------- per-step kernels ----------------

// Gated neighbor pooling: gated = sigmoid(zin@Wg+bg)*(zin@Wm); masked sum over
// 4 neighbor slots; Hin = (v==start_v) ? H[:,start_v] : Hsum.
// grid: (32 N-tiles, 2 dirs), block 128 (4 waves; wave w owns M-tiles 2w,2w+1)
__global__ __launch_bounds__(128) void k_pool(P p, int step) {
    const int dir     = blockIdx.y;
    const int v       = dir == 0 ? step : (MAXN - 1 - step);
    const int start_v = dir == 0 ? 0 : (MAXN - 1);
    const int ntile   = blockIdx.x;
    const int lane    = threadIdx.x & 31;
    const int wave    = threadIdx.x >> 5;
    const int g       = (lane >> 4) & 1;

    const int* idx = dir == 0 ? p.pred_idx : p.succ_idx;
    const int* et  = dir == 0 ? p.pred_et  : p.succ_et;
    const unsigned char* msk = dir == 0 ? p.pred_mask : p.succ_mask;
    const _Float16* H16  = p.H16[dir];
    const _Float16* Wg16 = p.Wg16[dir];
    const _Float16* Wm16 = p.Wm16[dir];

    __shared__ float lds_g[128 * 16];

    const int mt0 = wave * 2, mt1 = wave * 2 + 1;
    // per-fragment gather invariants (hoisted out of the K loop)
    const int m0 = mt0 * 16 + (lane & 15), m1 = mt1 * 16 + (lane & 15);
    const int b0 = m0 >> 2, s0 = m0 & 3, b1 = m1 >> 2, s1 = m1 & 3;
    const int u0 = idx[((b0 * MAXN) + v) * KNB + s0];
    const int u1 = idx[((b1 * MAXN) + v) * KNB + s1];
    const int e0 = et [((b0 * MAXN) + v) * KNB + s0];
    const int e1 = et [((b1 * MAXN) + v) * KNB + s1];
    const _Float16* h0 = H16 + ((size_t)b0 * MAXN + u0) * HSP;
    const _Float16* h1 = H16 + ((size_t)b1 * MAXN + u1) * HSP;

    v8f ag0 = {}, am0 = {}, ag1 = {}, am1 = {};

    for (int ks = 0; ks < KSTEPS; ++ks) {
        const int kbase = ks * 32;
        if (ks + 1 < KSTEPS) {
            __builtin_prefetch(&Wg16[(((size_t)ntile * KSTEPS + ks + 1) * 32) * 16], 0, 1);
            __builtin_prefetch(&Wm16[(((size_t)ntile * KSTEPS + ks + 1) * 32) * 16], 0, 1);
        }
        v16h a0 = load_A_row(h0, kbase, g);
        v16h a1 = load_A_row(h1, kbase, g);
        if (ks == KSTEPS - 1 && g == 0) {   // patch edge-type one-hot (cols 501,502)
            a0[13] = (_Float16)(e0 == 0 ? 1.0f : 0.0f);
            a0[14] = (_Float16)(e0 == 1 ? 1.0f : 0.0f);
            a1[13] = (_Float16)(e1 == 0 ? 1.0f : 0.0f);
            a1[14] = (_Float16)(e1 == 1 ? 1.0f : 0.0f);
        }
        v16h bg = load_B_sw(Wg16, ntile, ks, lane);
        v16h bm = load_B_sw(Wm16, ntile, ks, lane);
        ag0 = wmma_f16(a0, bg, ag0);
        am0 = wmma_f16(a0, bm, am0);
        ag1 = wmma_f16(a1, bg, ag1);
        am1 = wmma_f16(a1, bm, am1);
    }

    const int nl   = lane & 15;
    const int ncol = ntile * 16 + nl;
    const float bgc = (ncol < HS) ? p.bgv[dir][ncol] : 0.0f;

    const float mk0 = msk[((b0 * MAXN) + v) * KNB + s0] ? 1.0f : 0.0f;
    const float mk1 = msk[((b1 * MAXN) + v) * KNB + s1] ? 1.0f : 0.0f;
    // NOTE: lane holds C rows m = mt*16 + g*8 + r, which differ from the lane's
    // A-row; reload mask per C row instead of reusing mk0/mk1 blindly.
    (void)mk0; (void)mk1;
#pragma unroll
    for (int half = 0; half < 2; ++half) {
        const int mt = half ? mt1 : mt0;
        const v8f accg = half ? ag1 : ag0;
        const v8f accm = half ? am1 : am0;
#pragma unroll
        for (int r = 0; r < 8; ++r) {
            const int row = mt * 16 + g * 8 + r;        // (b*4 + slot)
            const int b = row >> 2, s = row & 3;
            const float mk = msk[((b * MAXN) + v) * KNB + s] ? 1.0f : 0.0f;
            lds_g[row * 16 + nl] = sigm(accg[r] + bgc) * accm[r] * mk;
        }
    }
    __syncthreads();

    for (int i = threadIdx.x; i < BB * 16; i += 128) {
        const int b = i >> 4, c = i & 15;
        const float hsum = lds_g[(b * 4 + 0) * 16 + c] + lds_g[(b * 4 + 1) * 16 + c]
                         + lds_g[(b * 4 + 2) * 16 + c] + lds_g[(b * 4 + 3) * 16 + c];
        const int n = ntile * 16 + c;
        const float hin = (v == start_v)
            ? p.H32[dir][((size_t)b * MAXN + v) * HSP + n] : hsum;
        p.Hin32[dir][b * HSP + n] = hin;
        p.Hin16[dir][b * HSP + n] = (_Float16)hin;
    }
}

// gh = Hin[32,HSP] @ whh.T[HSP,N3P]
// grid: (12, 2 dirs), block 128; wave w owns N-tiles blockIdx.x*8 + 2w, 2w+1
__global__ __launch_bounds__(128) void k_gh(P p, int step) {
    (void)step;
    const int dir  = blockIdx.y;
    const int lane = threadIdx.x & 31;
    const int wave = threadIdx.x >> 5;
    const int g    = (lane >> 4) & 1;
    const int nt0  = blockIdx.x * 8 + wave * 2;
    const int nt1  = nt0 + 1;

    const _Float16* A  = p.Hin16[dir];
    const _Float16* Bw = p.Whh16[dir];
    const _Float16* r0 = A + (size_t)(0 * 16 + (lane & 15)) * HSP;
    const _Float16* r1 = A + (size_t)(1 * 16 + (lane & 15)) * HSP;

    v8f c00 = {}, c01 = {}, c10 = {}, c11 = {};
    for (int ks = 0; ks < KSTEPS; ++ks) {
        const int kbase = ks * 32;
        if (ks + 1 < KSTEPS) {
            __builtin_prefetch(&Bw[(((size_t)nt0 * KSTEPS + ks + 1) * 32) * 16], 0, 1);
            __builtin_prefetch(&Bw[(((size_t)nt1 * KSTEPS + ks + 1) * 32) * 16], 0, 1);
        }
        v16h a0 = load_A_row(r0, kbase, g);
        v16h a1 = load_A_row(r1, kbase, g);
        v16h b0 = load_B_sw(Bw, nt0, ks, lane);
        v16h b1 = load_B_sw(Bw, nt1, ks, lane);
        c00 = wmma_f16(a0, b0, c00);
        c01 = wmma_f16(a0, b1, c01);
        c10 = wmma_f16(a1, b0, c10);
        c11 = wmma_f16(a1, b1, c11);
    }

    float* GH = p.GH[dir];
    const int nl = lane & 15;
#pragma unroll
    for (int r = 0; r < 8; ++r) {
        GH[(size_t)( 0 + g * 8 + r) * N3P + nt0 * 16 + nl] = c00[r];
        GH[(size_t)( 0 + g * 8 + r) * N3P + nt1 * 16 + nl] = c01[r];
        GH[(size_t)(16 + g * 8 + r) * N3P + nt0 * 16 + nl] = c10[r];
        GH[(size_t)(16 + g * 8 + r) * N3P + nt1 * 16 + nl] = c11[r];
    }
}

// GRU gate math; writes H[:,v] (f32 master + f16 mirror)
__global__ void k_combine(P p, int step) {
    const int dir = blockIdx.y;
    const int v   = dir == 0 ? step : (MAXN - 1 - step);
    const int i = blockIdx.x * blockDim.x + threadIdx.x;
    if (i >= BB * HS) return;
    const int b = i / HS, j = i % HS;

    const int ty = p.v_type[b * MAXN + v];
    const float* wih = p.wih[dir];
    const float* bih = p.bih[dir];
    const float* bhh = p.bhh[dir];
    const float* GH  = p.GH[dir] + (size_t)b * N3P;

    const float ir = wih[(j         ) * NVT + ty] + bih[j];
    const float iz = wih[(HS + j    ) * NVT + ty] + bih[HS + j];
    const float in = wih[(2 * HS + j) * NVT + ty] + bih[2 * HS + j];
    const float hr = GH[j]          + bhh[j];
    const float hz = GH[HS + j]     + bhh[HS + j];
    const float hn = GH[2 * HS + j] + bhh[2 * HS + j];

    const float r = sigm(ir + hr);
    const float z = sigm(iz + hz);
    const float n = tanhf(in + r * hn);
    const float h = p.Hin32[dir][b * HSP + j];
    const float hv = (1.0f - z) * n + z * h;

    p.H32[dir][((size_t)b * MAXN + v) * HSP + j] = hv;
    p.H16[dir][((size_t)b * MAXN + v) * HSP + j] = (_Float16)hv;
}

// ---------------- head (negligible FLOPs; plain f32) ----------------

__global__ void k_head1(P p) {
    const int i = blockIdx.x * blockDim.x + threadIdx.x;
    if (i >= BB * HS) return;
    const int b = i / HS, n = i % HS;
    const float* hf = p.H32[0] + ((size_t)b * MAXN + (MAXN - 1)) * HSP;
    const float* hb = p.H32[1] + ((size_t)b * MAXN + 0) * HSP;
    float acc = p.bu[n];
    for (int k = 0; k < HS; ++k) acc += hf[k] * p.Wu[(size_t)k * HS + n];
    for (int k = 0; k < HS; ++k) acc += hb[k] * p.Wu[(size_t)(HS + k) * HS + n];
    p.Hg[b * HS + n] = acc;
}

__global__ void k_bn(float* X, const float* gamma, const float* beta, int ncols) {
    const int n = blockIdx.x * blockDim.x + threadIdx.x;
    if (n >= ncols) return;
    float m = 0.0f, s = 0.0f;
    for (int b = 0; b < BB; ++b) { const float x = X[b * ncols + n]; m += x; s += x * x; }
    m *= (1.0f / BB);
    s = s * (1.0f / BB) - m * m;
    const float inv = rsqrtf(s + 1e-5f);
    for (int b = 0; b < BB; ++b)
        X[b * ncols + n] = gamma[n] * (X[b * ncols + n] - m) * inv + beta[n];
}

__global__ void k_head2(P p) {
    const int i = blockIdx.x * blockDim.x + threadIdx.x;
    if (i >= BB * 2 * HS) return;
    const int b = i / (2 * HS), j = i % (2 * HS);
    float acc = p.bc1[j];
    for (int n = 0; n < HS; ++n) acc += p.Hg[b * HS + n] * p.Wc1[(size_t)n * (2 * HS) + j];
    p.h1[b * (2 * HS) + j] = fmaxf(acc, 0.0f);
}

__global__ void k_head3(P p, float* out) {
    const int b = threadIdx.x;
    if (b >= BB) return;
    float acc = p.bc2v[0];
    for (int j = 0; j < 2 * HS; ++j) acc += p.h1[b * (2 * HS) + j] * p.Wc2[j];
    out[b] = acc;
}

// ---------------- launch ----------------

extern "C" void kernel_launch(void* const* d_in, const int* in_sizes, int n_in,
                              void* d_out, int out_size, void* d_ws, size_t ws_size,
                              hipStream_t stream) {
    (void)in_sizes; (void)n_in; (void)out_size; (void)ws_size;
    P p{};
    p.v_type    = (const int*)d_in[0];
    p.pred_idx  = (const int*)d_in[1];
    p.pred_mask = (const unsigned char*)d_in[2];
    p.pred_et   = (const int*)d_in[3];
    p.succ_idx  = (const int*)d_in[4];
    p.succ_mask = (const unsigned char*)d_in[5];
    p.succ_et   = (const int*)d_in[6];
    p.wih[0] = (const float*)d_in[7];  p.whh[0] = (const float*)d_in[8];
    p.bih[0] = (const float*)d_in[9];  p.bhh[0] = (const float*)d_in[10];
    p.wih[1] = (const float*)d_in[11]; p.whh[1] = (const float*)d_in[12];
    p.bih[1] = (const float*)d_in[13]; p.bhh[1] = (const float*)d_in[14];
    p.Wg[0]  = (const float*)d_in[15]; p.bgv[0] = (const float*)d_in[16];
    p.Wm[0]  = (const float*)d_in[17];
    p.Wg[1]  = (const float*)d_in[18]; p.bgv[1] = (const float*)d_in[19];
    p.Wm[1]  = (const float*)d_in[20];
    p.Wu  = (const float*)d_in[21]; p.bu  = (const float*)d_in[22];
    p.gu  = (const float*)d_in[23]; p.beu = (const float*)d_in[24];
    p.Wc1 = (const float*)d_in[25]; p.bc1 = (const float*)d_in[26];
    p.gc  = (const float*)d_in[27]; p.bec = (const float*)d_in[28];
    p.Wc2 = (const float*)d_in[29]; p.bc2v = (const float*)d_in[30];

    // carve workspace
    char* base = (char*)d_ws;
    size_t off = 0;
    auto carve = [&](size_t bytes) -> char* {
        char* ptr = base + off;
        off = (off + bytes + 255) & ~(size_t)255;
        return ptr;
    };
    for (int d = 0; d < 2; ++d) p.H32[d]   = (float*)   carve((size_t)BB * MAXN * HSP * 4);
    for (int d = 0; d < 2; ++d) p.H16[d]   = (_Float16*)carve((size_t)BB * MAXN * HSP * 2);
    for (int d = 0; d < 2; ++d) p.Wg16[d]  = (_Float16*)carve((size_t)HSP * HSP * 2);
    for (int d = 0; d < 2; ++d) p.Wm16[d]  = (_Float16*)carve((size_t)HSP * HSP * 2);
    for (int d = 0; d < 2; ++d) p.Whh16[d] = (_Float16*)carve((size_t)HSP * N3P * 2);
    for (int d = 0; d < 2; ++d) p.Hin32[d] = (float*)   carve((size_t)BB * HSP * 4);
    for (int d = 0; d < 2; ++d) p.Hin16[d] = (_Float16*)carve((size_t)BB * HSP * 2);
    for (int d = 0; d < 2; ++d) p.GH[d]    = (float*)   carve((size_t)BB * N3P * 4);
    p.Hg = (float*)carve((size_t)BB * HS * 4);
    p.h1 = (float*)carve((size_t)BB * 2 * HS * 4);

    k_init<<<1024, 256, 0, stream>>>(p);
    k_prep<<<2048, 256, 0, stream>>>(p);

    const int combine_blocks = (BB * HS + 255) / 256;      // 63
    for (int round = 0; round <= NROUNDS; ++round) {       // 27 sweeps
        for (int step = 0; step < MAXN; ++step) {
            k_pool   <<<dim3(HSP / 16, 2), 128, 0, stream>>>(p, step);
            k_gh     <<<dim3(N3P / 16 / 8, 2), 128, 0, stream>>>(p, step);
            k_combine<<<dim3(combine_blocks, 2), 256, 0, stream>>>(p, step);
        }
    }

    k_head1<<<(BB * HS + 255) / 256, 256, 0, stream>>>(p);
    k_bn   <<<(HS + 255) / 256, 256, 0, stream>>>(p.Hg, p.gu, p.beu, HS);
    k_head2<<<(BB * 2 * HS + 255) / 256, 256, 0, stream>>>(p);
    k_bn   <<<(2 * HS + 255) / 256, 256, 0, stream>>>(p.h1, p.gc, p.bec, 2 * HS);
    k_head3<<<1, 32, 0, stream>>>(p, (float*)d_out);
}
